// MultiHeadCore_46213848106010
// MI455X (gfx1250) — matrix operand order repaired and checked by reading, not compiled, not run on hardware
//
#include <hip/hip_runtime.h>
#include <hip/hip_bf16.h>

typedef __attribute__((ext_vector_type(16))) __bf16 v16bf;
typedef __attribute__((ext_vector_type(8)))  float  v8f;

#define NB 16
#define NN 256
#define DD 512
#define NH 8
#define DK 64

// ---- WMMA fragment layout helpers (CDNA5 ISA 7.12.2, wave32) ----
// A-matrix 16x32 bf16: lane L: M = L%16, half = L/16.
// element e: K = half*16 + e (contiguous order used by this code, B matches; not the ISA table order)
__device__ __forceinline__ int fragA_off(int mm, int kk) {
    int half = kk >> 4;
    int e    = kk & 15;
    return (mm + half * 16) * 16 + e;
}
// B-matrix 32x16 bf16: lane = (n%16) + 16*(k/16), element e = k%16
__device__ __forceinline__ int fragB_off(int kk, int nn) {
    return ((nn & 15) + ((kk >> 4) << 4)) * 16 + (kk & 15);
}

// generic LDS pointer -> wave-relative LDS byte offset (low 32 bits of the
// shared aperture address, per ISA 10.2 aperture mapping)
__device__ __forceinline__ uint32_t lds_off32(const void* p) {
    return (uint32_t)(uintptr_t)p;
}
// async 16-byte global->LDS copy (ASYNCcnt), no VGPR data round-trip
__device__ __forceinline__ void async_b128(uint32_t lds, const void* g) {
    asm volatile("global_load_async_to_lds_b128 %0, %1, off"
                 :: "v"(lds), "v"(g) : "memory");
}
__device__ __forceinline__ void wait_async0() {
    asm volatile("s_wait_asynccnt 0x0" ::: "memory");
}

// =================================================================
// Kernel 1: geometric embedding -> log(clip(relu(wg))) per head (bf16)
// =================================================================
__global__ __launch_bounds__(256)
void geo_logwg_kernel(const float* __restrict__ box,
                      const float* __restrict__ Wg,
                      const float* __restrict__ bg,
                      __bf16* __restrict__ logwg) {
    __shared__ float sWg[NH * 64];
    __shared__ float sBg[NH];
    int tid = threadIdx.x;
    for (int q = tid; q < NH * 64; q += 256) sWg[q] = Wg[q];
    if (tid < NH) sBg[tid] = bg[tid];
    __syncthreads();

    size_t idx = (size_t)blockIdx.x * 256 + tid;
    int j = (int)(idx & (NN - 1));
    int i = (int)((idx >> 8) & (NN - 1));
    int b = (int)(idx >> 16);

    const float* bi = box + ((size_t)b * NN + i) * 4;
    const float* bj = box + ((size_t)b * NN + j) * 4;
    float cxi = (bi[0] + bi[2]) * 0.5f, cyi = (bi[1] + bi[3]) * 0.5f;
    float wi  = bi[2] - bi[0] + 1.0f,   hi  = bi[3] - bi[1] + 1.0f;
    float cxj = (bj[0] + bj[2]) * 0.5f, cyj = (bj[1] + bj[3]) * 0.5f;
    float wj  = bj[2] - bj[0] + 1.0f,   hj  = bj[3] - bj[1] + 1.0f;

    float pos[4];
    pos[0] = __logf(fmaxf(fabsf((cxi - cxj) / wi), 1e-3f));
    pos[1] = __logf(fmaxf(fabsf((cyi - cyj) / hi), 1e-3f));
    pos[2] = __logf(wi / wj);
    pos[3] = __logf(hi / hj);

    float sv[32], cv[32];
#pragma unroll
    for (int c = 0; c < 8; ++c) {
        float freq = 100.0f * __expf(-(float)c * (0.125f * 6.907755278982137f));
#pragma unroll
        for (int d = 0; d < 4; ++d) {
            float m = pos[d] * freq;
            int t = d * 8 + c;
            sv[t] = __sinf(m);
            cv[t] = __cosf(m);
        }
    }
#pragma unroll
    for (int h = 0; h < NH; ++h) {
        float acc = sBg[h];
#pragma unroll
        for (int t = 0; t < 32; ++t)
            acc += sv[t] * sWg[h * 64 + t] + cv[t] * sWg[h * 64 + 32 + t];
        float lg = __logf(fmaxf(acc, 1e-6f));
        logwg[(((size_t)b * NH + h) * NN + i) * NN + j] = (__bf16)lg;
    }
}

// =================================================================
// Kernel 2: GEMM + bias.  OMODE: 0 = f32 row-major, 1 = bf16 row-major,
//                         2 = bf16 per-head-transposed [B,H,DK,NN] (for V)
// =================================================================
template<int OMODE>
__global__ __launch_bounds__(256)
void gemm_bias_kernel(const float* __restrict__ X, const float* __restrict__ W,
                      const float* __restrict__ bias, float* __restrict__ Yf,
                      __bf16* __restrict__ Yb, int Kdim, int Ncols) {
    __shared__ __bf16 sA[4 * 512];
    __shared__ __bf16 sB[4 * 512];
    int tid = threadIdx.x;
    int lane = tid & 31;
    int wave = tid >> 5;
    int rowBase = blockIdx.x * 64;
    int colBase = blockIdx.y * 64;
    int mi = wave >> 1;
    int nh = wave & 1;

    v8f acc0 = {}; v8f acc1 = {};
    int nKc = Kdim >> 5;
    for (int kc = 0; kc < nKc; ++kc) {
        int kBase = kc << 5;
        // prefetch next k-chunk into caches (global_prefetch_b8)
        if (kc + 1 < nKc) {
            __builtin_prefetch(&X[(size_t)(rowBase + (tid & 63)) * Kdim + kBase + 32], 0, 0);
            __builtin_prefetch(&W[(size_t)(kBase + 32 + (tid & 31)) * Ncols + colBase], 0, 0);
        }
        for (int q = tid; q < 2048; q += 256) {
            int m = q >> 5, kk = q & 31;
            float v = X[(size_t)(rowBase + m) * Kdim + kBase + kk];
            sA[(m >> 4) * 512 + fragA_off(m & 15, kk)] = (__bf16)v;
        }
        for (int q = tid; q < 2048; q += 256) {
            int kk = q >> 6, n = q & 63;
            float v = W[(size_t)(kBase + kk) * Ncols + colBase + n];
            sB[(n >> 4) * 512 + fragB_off(kk, n & 15)] = (__bf16)v;
        }
        __syncthreads();
        v16bf a  = *(const v16bf*)(sA + mi * 512 + lane * 16);
        v16bf b0 = *(const v16bf*)(sB + (nh * 2 + 0) * 512 + lane * 16);
        v16bf b1 = *(const v16bf*)(sB + (nh * 2 + 1) * 512 + lane * 16);
        acc0 = __builtin_amdgcn_wmma_f32_16x16x32_bf16(false, a, false, b0,
                                                       (short)0, acc0, false, false);
        acc1 = __builtin_amdgcn_wmma_f32_16x16x32_bf16(false, a, false, b1,
                                                       (short)0, acc1, false, false);
        __syncthreads();
    }
    int Mm = 8 * (lane >> 4);
    int nn = lane & 15;
#pragma unroll
    for (int r = 0; r < 8; ++r) {
        int row = rowBase + mi * 16 + Mm + r;
        int c0  = colBase + (nh * 2 + 0) * 16 + nn;
        int c1  = colBase + (nh * 2 + 1) * 16 + nn;
        float y0 = acc0[r] + bias[c0];
        float y1 = acc1[r] + bias[c1];
        if (OMODE == 0) {
            Yf[(size_t)row * Ncols + c0] = y0;
            Yf[(size_t)row * Ncols + c1] = y1;
        } else if (OMODE == 1) {
            Yb[(size_t)row * Ncols + c0] = (__bf16)y0;
            Yb[(size_t)row * Ncols + c1] = (__bf16)y1;
        } else {
            // row = b*256 + token, col = h*64 + dk  ->  [B,H,DK,NN]
            int bb = row >> 8, ntok = row & 255;
            Yb[((((size_t)bb * NH + (c0 >> 6)) * DK + (c0 & 63)) << 8) + ntok] = (__bf16)y0;
            Yb[((((size_t)bb * NH + (c1 >> 6)) * DK + (c1 & 63)) << 8) + ntok] = (__bf16)y1;
        }
    }
}

// =================================================================
// Kernel 3: attention core per (b, h, 64-row chunk)
//   staging via GLOBAL_LOAD_ASYNC_TO_LDS_B128 (bf16 inputs, frag order)
// =================================================================
__global__ __launch_bounds__(256)
void attn_kernel(const __bf16* __restrict__ qb, const __bf16* __restrict__ kb,
                 const __bf16* __restrict__ vbT, const __bf16* __restrict__ logwg,
                 float* __restrict__ attn_out) {
    extern __shared__ char smem[];
    __bf16* sQ  = (__bf16*)(smem);                  //  8 KB: Q 64x64 (A frags)
    __bf16* sKT = (__bf16*)(smem + 8192);           // 32 KB: K^T 64x256 (B frags)
    __bf16* sV  = (__bf16*)(smem + 8192 + 32768);   // 32 KB: V 256x64 (B frags)
    float*  sS  = (float*)(smem + 8192 + 65536);    // 64 KB: S 64x256 f32
    __bf16* sP  = sKT;                              // reuse after GEMM1

    int tid = threadIdx.x, lane = tid & 31, wave = tid >> 5;
    int m0 = blockIdx.x * 64;
    int h  = blockIdx.y;
    int b  = blockIdx.z;

    uint32_t qBase = lds_off32(sQ), kBase = lds_off32(sKT), vBase = lds_off32(sV);

    // ---- Q [64x64] -> A-frag order: 512 chunks of 8 bf16 ----
    for (int c = tid; c < 512; c += 256) {
        int m = c >> 3, kg = c & 7;                 // kg: 8-elem k-group
        int g2 = kg & 3;
        int frag = (m >> 4) * 2 + (kg >> 2);
        uint32_t loff = qBase + (uint32_t)(frag * 1024 +
                        ((m & 15) + (g2 >> 1) * 16) * 32 + (g2 & 1) * 16);
        async_b128(loff, qb + ((size_t)b * NN + m0 + m) * DD + h * DK + kg * 8);
    }
    // ---- K^T (kdim 64, n 256) -> B-frag order: 2048 chunks ----
    for (int c = tid; c < 2048; c += 256) {
        int n = c >> 3, kc = (c >> 1) & 3, hb = c & 1;
        int frag = (kc >> 1) * 16 + (n >> 4);
        uint32_t loff = kBase + (uint32_t)(frag * 1024 +
                        ((n & 15) + (kc & 1) * 16) * 32 + hb * 16);
        async_b128(loff, kb + ((size_t)b * NN + n) * DD + h * DK + kc * 16 + hb * 8);
    }
    // ---- V (kv 256, n 64) from vbT[B,H,DK,NN] -> B-frag order: 2048 chunks ----
    for (int c = tid; c < 2048; c += 256) {
        int n = c >> 5, kvc = (c >> 1) & 15, hb = c & 1;
        int frag = (kvc >> 1) * 4 + (n >> 4);
        uint32_t loff = vBase + (uint32_t)(frag * 1024 +
                        ((n & 15) + (kvc & 1) * 16) * 32 + hb * 16);
        async_b128(loff, vbT + (((size_t)b * NH + h) * DK + n) * NN + kvc * 16 + hb * 8);
    }
    wait_async0();
    __syncthreads();

    // ---- GEMM1: S[64 x 256] = Q K^T ----
    {
        int mi = wave >> 1, nh = wave & 1;
        v16bf a0 = *(const v16bf*)(sQ + (mi * 2 + 0) * 512 + lane * 16);
        v16bf a1 = *(const v16bf*)(sQ + (mi * 2 + 1) * 512 + lane * 16);
        int Mm = 8 * (lane >> 4), nn = lane & 15;
        for (int nt8 = 0; nt8 < 8; ++nt8) {
            int nt = nh * 8 + nt8;
            v8f acc = {};
            v16bf b0 = *(const v16bf*)(sKT + (0 * 16 + nt) * 512 + lane * 16);
            v16bf b1 = *(const v16bf*)(sKT + (1 * 16 + nt) * 512 + lane * 16);
            acc = __builtin_amdgcn_wmma_f32_16x16x32_bf16(false, a0, false, b0,
                                                          (short)0, acc, false, false);
            acc = __builtin_amdgcn_wmma_f32_16x16x32_bf16(false, a1, false, b1,
                                                          (short)0, acc, false, false);
#pragma unroll
            for (int r = 0; r < 8; ++r)
                sS[(mi * 16 + Mm + r) * 256 + nt * 16 + nn] = acc[r];
        }
    }
    __syncthreads();

    // ---- softmax rows (fused logwg bias), one row per thread ----
    if (tid < 64) {
        int row = tid;
        const float scale = 0.125f; // 1/sqrt(DK)
        const __bf16* lw = logwg + (((size_t)b * NH + h) * NN + m0 + row) * NN;
        float* srow = sS + row * 256;
        float mx = -1e30f;
        for (int j = 0; j < 256; ++j) {
            float v = srow[j] * scale + (float)lw[j];
            srow[j] = v;
            mx = fmaxf(mx, v);
        }
        float sum = 0.f;
        for (int j = 0; j < 256; ++j) {
            float e = __expf(srow[j] - mx);
            srow[j] = e;
            sum += e;
        }
        float inv = 1.0f / sum;
        int mi = row >> 4, mm = row & 15;
        for (int j = 0; j < 256; ++j)
            sP[(mi * 8 + (j >> 5)) * 512 + fragA_off(mm, j & 31)] =
                (__bf16)(srow[j] * inv);
    }
    __syncthreads();

    // ---- GEMM2: out[64 x 64] = P V ----
    {
        int mi = wave >> 1, nh = wave & 1;
        int Mm = 8 * (lane >> 4), nn = lane & 15;
        for (int nt2 = 0; nt2 < 2; ++nt2) {
            int nt = nh * 2 + nt2;
            v8f acc = {};
#pragma unroll
            for (int kt = 0; kt < 8; ++kt) {
                v16bf a  = *(const v16bf*)(sP + (mi * 8 + kt) * 512 + lane * 16);
                v16bf bb = *(const v16bf*)(sV + (kt * 4 + nt) * 512 + lane * 16);
                acc = __builtin_amdgcn_wmma_f32_16x16x32_bf16(false, a, false, bb,
                                                              (short)0, acc, false, false);
            }
#pragma unroll
            for (int r = 0; r < 8; ++r)
                attn_out[((size_t)b * NN + m0 + mi * 16 + Mm + r) * DD +
                         h * DK + nt * 16 + nn] = acc[r];
        }
    }
}

// =================================================================
extern "C" void kernel_launch(void* const* d_in, const int* in_sizes, int n_in,
                              void* d_out, int out_size, void* d_ws, size_t ws_size,
                              hipStream_t stream) {
    (void)in_sizes; (void)n_in; (void)out_size; (void)ws_size;
    const float* q_in = (const float*)d_in[0];
    const float* k_in = (const float*)d_in[1];
    const float* v_in = (const float*)d_in[2];
    const float* box  = (const float*)d_in[3];
    const float* Wq   = (const float*)d_in[4];
    const float* bq   = (const float*)d_in[5];
    const float* Wk   = (const float*)d_in[6];
    const float* bk   = (const float*)d_in[7];
    const float* Wv   = (const float*)d_in[8];
    const float* bv   = (const float*)d_in[9];
    const float* Wo   = (const float*)d_in[10];
    const float* bo   = (const float*)d_in[11];
    const float* Wg   = (const float*)d_in[12];
    const float* bg   = (const float*)d_in[13];

    char* ws = (char*)d_ws;
    __bf16* qb    = (__bf16*)(ws);                       // 4 MB [B,N,D] bf16
    __bf16* kb    = (__bf16*)(ws +  4ull * 1024 * 1024); // 4 MB [B,N,D] bf16
    __bf16* vbT   = (__bf16*)(ws +  8ull * 1024 * 1024); // 4 MB [B,H,DK,N] bf16
    float*  attn  = (float*)(ws + 12ull * 1024 * 1024);  // 8 MB [B,N,D] f32
    __bf16* logwg = (__bf16*)(ws + 20ull * 1024 * 1024); // 16 MB [B,H,N,N] bf16

    geo_logwg_kernel<<<(NB * NN * NN) / 256, 256, 0, stream>>>(box, Wg, bg, logwg);

    dim3 ggrid(NB * NN / 64, DD / 64);
    gemm_bias_kernel<1><<<ggrid, 256, 0, stream>>>(q_in, Wq, bq, nullptr, qb, DD, DD);
    gemm_bias_kernel<1><<<ggrid, 256, 0, stream>>>(k_in, Wk, bk, nullptr, kb, DD, DD);
    gemm_bias_kernel<2><<<ggrid, 256, 0, stream>>>(v_in, Wv, bv, nullptr, vbT, DD, DD);

    size_t smem = 8192 + 32768 + 32768 + 65536; // 136 KB
    hipFuncSetAttribute(reinterpret_cast<const void*>(attn_kernel),
                        hipFuncAttributeMaxDynamicSharedMemorySize, (int)smem);
    attn_kernel<<<dim3(NN / 64, NH, NB), 256, smem, stream>>>(qb, kb, vbT,
                                                              logwg, attn);

    gemm_bias_kernel<0><<<ggrid, 256, 0, stream>>>(attn, Wo, bo, (float*)d_out,
                                                   nullptr, DD, DD);
}
